// SHGN_23888608100650
// MI455X (gfx1250) — matrix operand order repaired and checked by reading, not compiled, not run on hardware
//
#include <hip/hip_runtime.h>
#include <hip/hip_bf16.h>
#include <math.h>

// ---------------------------------------------------------------------------
// SHGN (3-layer heterogeneous GAT) for MI455X / gfx1250, wave32.
//  - GEMM phase: bf16 WMMA (v_wmma_f32_16x16x32_bf16), wave -> 16x32 tile.
//  - Edge phase: bandwidth-bound gather/scatter kernels (L2-resident).
// ---------------------------------------------------------------------------

typedef __attribute__((ext_vector_type(16))) __bf16 bf16x16;
typedef __attribute__((ext_vector_type(8)))  __bf16 bf16x8;
typedef __attribute__((ext_vector_type(8)))  float  v8f;

#define NEG_SLOPE 0.2f

// ---------------- monotone float<->uint key for atomicMax on floats --------
__device__ __forceinline__ unsigned fkey(float f) {
    unsigned u = __float_as_uint(f);
    return (u & 0x80000000u) ? ~u : (u | 0x80000000u);
}
__device__ __forceinline__ float funkey(unsigned k) {
    unsigned u = (k & 0x80000000u) ? (k & 0x7fffffffu) : ~k;
    return __uint_as_float(u);
}

// ---------------- fp32 -> bf16 conversion ----------------------------------
__global__ void cvt_f32_bf16(const float* __restrict__ in,
                             __bf16* __restrict__ out, int n) {
    int i = blockIdx.x * blockDim.x + threadIdx.x;
    if (i < n) out[i] = (__bf16)in[i];
}

// ---------------- WMMA GEMM: C[M,Nc] = A[M,K] (bf16) @ W[Nc,K]^T (bf16) ----
// One wave computes a 16x32 output strip (2 accumulators, shared A fragment).
__global__ void gemm_wmma_bf16(const __bf16* __restrict__ A,
                               const __bf16* __restrict__ W,
                               float* __restrict__ C,
                               int M, int K, int Nc) {
    const int lane  = threadIdx.x & 31;
    const int wave  = threadIdx.x >> 5;
    const int nSup  = Nc >> 5;                       // 32-column supertiles
    const int tiles = (M >> 4) * nSup;
    const int t = blockIdx.x * (blockDim.x >> 5) + wave;
    if (t >= tiles) return;                          // whole-wave exit: EXEC stays all-1s

    const int tm  = t / nSup;
    const int tn  = t % nSup;
    const int l15 = lane & 15;
    const int m   = (tm << 4) + l15;
    const int kbA = (lane < 16) ? 0 : 16;            // A: contiguous k = 16*(lane/16) + s
    const int kbB = (lane < 16) ? 0 : 16;            // B: contiguous k = 16*(lane/16) + s, same as A
    const __bf16* arow = A + (size_t)m * K;
    const __bf16* w0   = W + (size_t)((tn << 5) + l15) * K;        // B col n0
    const __bf16* w1   = W + (size_t)((tn << 5) + 16 + l15) * K;   // B col n0+16

    v8f acc0 = {}; v8f acc1 = {};
    for (int k0 = 0; k0 < K; k0 += 32) {
        bf16x8 alo = *(const bf16x8*)(arow + k0 + kbA);
        bf16x8 ahi = *(const bf16x8*)(arow + k0 + kbA + 8);
        bf16x16 a;
#pragma unroll
        for (int i = 0; i < 8; ++i) { a[i] = alo[i]; a[i + 8] = ahi[i]; }

        bf16x8 b0l = *(const bf16x8*)(w0 + k0 + kbB);
        bf16x8 b0h = *(const bf16x8*)(w0 + k0 + kbB + 8);
        bf16x16 b0;
#pragma unroll
        for (int i = 0; i < 8; ++i) { b0[i] = b0l[i]; b0[i + 8] = b0h[i]; }

        bf16x8 b1l = *(const bf16x8*)(w1 + k0 + kbB);
        bf16x8 b1h = *(const bf16x8*)(w1 + k0 + kbB + 8);
        bf16x16 b1;
#pragma unroll
        for (int i = 0; i < 8; ++i) { b1[i] = b1l[i]; b1[i + 8] = b1h[i]; }

        acc0 = __builtin_amdgcn_wmma_f32_16x16x32_bf16(
                   false, a, false, b0, (short)0, acc0, false, false);
        acc1 = __builtin_amdgcn_wmma_f32_16x16x32_bf16(
                   false, a, false, b1, (short)0, acc1, false, false);
    }

    // C/D layout: lanes 0-15 -> rows 0..7, lanes 16-31 -> rows 8..15
    const int mrow = (tm << 4) + ((lane >> 4) << 3);
    const int ncol = (tn << 5) + l15;
#pragma unroll
    for (int r = 0; r < 8; ++r) {
        C[(size_t)(mrow + r) * Nc + ncol]      = acc0[r];
        C[(size_t)(mrow + r) * Nc + ncol + 16] = acc1[r];
    }
}

// ---------------- per-etype attention table ee[ET,H] -----------------------
// ee[et,h] = sum_j ae[h,j] * sum_k Ee[et,k] * We[h*64+j, k]   (D = De = 64)
__global__ void ee_table(const float* __restrict__ Ee, const float* __restrict__ We,
                         const float* __restrict__ ae, float* __restrict__ ee, int ET) {
    int idx = threadIdx.x;
    if (idx >= ET * 4) return;
    int et = idx >> 2, h = idx & 3;
    float s = 0.f;
    for (int j = 0; j < 64; ++j) {
        const float* wrow = We + (size_t)(h * 64 + j) * 64;
        float ef = 0.f;
        for (int k = 0; k < 64; ++k) ef += Ee[et * 64 + k] * wrow[k];
        s += ef * ae[h * 64 + j];
    }
    ee[idx] = s;
}

// ---------------- per-node attention dots aL/aR [N,H] -----------------------
__global__ void node_attn_dots(const float* __restrict__ Xp,
                               const float* __restrict__ al,
                               const float* __restrict__ ar,
                               float* __restrict__ aL, float* __restrict__ aR,
                               int N, int F) {
    int idx = blockIdx.x * blockDim.x + threadIdx.x;   // over N*H
    if (idx >= N * 4) return;
    int n = idx >> 2, h = idx & 3;
    const float* xv  = Xp + (size_t)n * (4 * F) + h * F;
    const float* alv = al + h * F;
    const float* arv = ar + h * F;
    float sl = 0.f, sr = 0.f;
    for (int f = 0; f < F; ++f) { sl += xv[f] * alv[f]; sr += xv[f] * arv[f]; }
    aL[idx] = sl; aR[idx] = sr;
}

// ---------------- edge pass 1: leaky-relu scores + segment max -------------
__global__ void edge_scores(const int* __restrict__ row, const int* __restrict__ col,
                            const int* __restrict__ et,
                            const float* __restrict__ aL, const float* __restrict__ aR,
                            const float* __restrict__ ee,
                            float* __restrict__ sbuf, unsigned* __restrict__ smaxkey,
                            int E) {
    int stride = gridDim.x * blockDim.x;
    for (int e = blockIdx.x * blockDim.x + threadIdx.x; e < E; e += stride) {
        int r = row[e], c = col[e], t = et[e];
        float4 l4 = ((const float4*)aL)[r];
        float4 r4 = ((const float4*)aR)[c];
        float4 e4 = ((const float4*)ee)[t];
        float s0 = l4.x + r4.x + e4.x, s1 = l4.y + r4.y + e4.y;
        float s2 = l4.z + r4.z + e4.z, s3 = l4.w + r4.w + e4.w;
        s0 = s0 > 0.f ? s0 : NEG_SLOPE * s0;
        s1 = s1 > 0.f ? s1 : NEG_SLOPE * s1;
        s2 = s2 > 0.f ? s2 : NEG_SLOPE * s2;
        s3 = s3 > 0.f ? s3 : NEG_SLOPE * s3;
        ((float4*)sbuf)[e] = make_float4(s0, s1, s2, s3);
        atomicMax(&smaxkey[c * 4 + 0], fkey(s0));
        atomicMax(&smaxkey[c * 4 + 1], fkey(s1));
        atomicMax(&smaxkey[c * 4 + 2], fkey(s2));
        atomicMax(&smaxkey[c * 4 + 3], fkey(s3));
    }
}

// ---------------- edge pass 2: p = exp(s - max), segment sum ---------------
__global__ void edge_expsum(const int* __restrict__ col,
                            float* __restrict__ sbuf,
                            const unsigned* __restrict__ smaxkey,
                            float* __restrict__ denom, int E) {
    int stride = gridDim.x * blockDim.x;
    for (int e = blockIdx.x * blockDim.x + threadIdx.x; e < E; e += stride) {
        int c = col[e];
        float4 s4 = ((const float4*)sbuf)[e];
        float p0 = __expf(s4.x - funkey(smaxkey[c * 4 + 0]));
        float p1 = __expf(s4.y - funkey(smaxkey[c * 4 + 1]));
        float p2 = __expf(s4.z - funkey(smaxkey[c * 4 + 2]));
        float p3 = __expf(s4.w - funkey(smaxkey[c * 4 + 3]));
        ((float4*)sbuf)[e] = make_float4(p0, p1, p2, p3);
        atomicAdd(&denom[c * 4 + 0], p0);
        atomicAdd(&denom[c * 4 + 1], p1);
        atomicAdd(&denom[c * 4 + 2], p2);
        atomicAdd(&denom[c * 4 + 3], p3);
    }
}

// ---------------- edge pass 3: weighted scatter-aggregate ------------------
// One wave per edge; lanes stripe the H*F features. Prefetch next source row.
__global__ void edge_scatter(const int* __restrict__ row, const int* __restrict__ col,
                             const float* __restrict__ sbuf,
                             const float* __restrict__ denom,
                             const float* __restrict__ Xp,
                             float* __restrict__ out, int E, int F) {
    const int HF   = 4 * F;
    const int lane = threadIdx.x & 31;
    const int wave = (blockIdx.x * blockDim.x + threadIdx.x) >> 5;
    const int nw   = (gridDim.x * blockDim.x) >> 5;
    for (int e = wave; e < E; e += nw) {
        int r = row[e], c = col[e];
        int en = e + nw;
        if (en < E) {                     // pull next source row toward L2/L0
            int rn = row[en];
            __builtin_prefetch(Xp + (size_t)rn * HF + lane * 8, 0, 1);
        }
        float4 p4 = ((const float4*)sbuf)[e];
        float4 d4 = ((const float4*)denom)[c];
        float attn[4];
        attn[0] = p4.x / fmaxf(d4.x, 1e-38f);
        attn[1] = p4.y / fmaxf(d4.y, 1e-38f);
        attn[2] = p4.z / fmaxf(d4.z, 1e-38f);
        attn[3] = p4.w / fmaxf(d4.w, 1e-38f);
        const float* xr = Xp + (size_t)r * HF;
        float* oc = out + (size_t)c * HF;
        for (int j = lane; j < HF; j += 32)
            atomicAdd(&oc[j], xr[j] * attn[j / F]);
    }
}

// ---------------- elementwise relu -----------------------------------------
__global__ void relu_k(float* __restrict__ x, int n) {
    int i = blockIdx.x * blockDim.x + threadIdx.x;
    if (i < n) x[i] = fmaxf(x[i], 0.f);
}

// ---------------- final: head-mean + L2 normalize --------------------------
__global__ void final_norm(const float* __restrict__ out /*[N,4*16]*/,
                           float* __restrict__ logits /*[N,16]*/, int N) {
    int n = blockIdx.x * blockDim.x + threadIdx.x;
    if (n >= N) return;
    const float* o = out + (size_t)n * 64;
    float v[16]; float ss = 0.f;
#pragma unroll
    for (int c = 0; c < 16; ++c) {
        float m = 0.25f * (o[c] + o[16 + c] + o[32 + c] + o[48 + c]);
        v[c] = m; ss += m * m;
    }
    float inv = 1.0f / fmaxf(sqrtf(ss), 1e-12f);
#pragma unroll
    for (int c = 0; c < 16; ++c) logits[(size_t)n * 16 + c] = v[c] * inv;
}

// ===========================================================================
extern "C" void kernel_launch(void* const* d_in, const int* in_sizes, int n_in,
                              void* d_out, int out_size, void* d_ws, size_t ws_size,
                              hipStream_t stream) {
    const float* x   = (const float*)d_in[0];
    const int*   row = (const int*)d_in[1];
    const int*   col = (const int*)d_in[2];
    const int*   et  = (const int*)d_in[3];

    const int N  = in_sizes[0] / 64;
    const int E  = in_sizes[1];
    const int ET = in_sizes[6] / 64;

    const float* Wl[3]  = { (const float*)d_in[4],  (const float*)d_in[10], (const float*)d_in[16] };
    const float* Wel[3] = { (const float*)d_in[5],  (const float*)d_in[11], (const float*)d_in[17] };
    const float* Eel[3] = { (const float*)d_in[6],  (const float*)d_in[12], (const float*)d_in[18] };
    const float* all[3] = { (const float*)d_in[7],  (const float*)d_in[13], (const float*)d_in[19] };
    const float* arl[3] = { (const float*)d_in[8],  (const float*)d_in[14], (const float*)d_in[20] };
    const float* ael[3] = { (const float*)d_in[9],  (const float*)d_in[15], (const float*)d_in[21] };

    const int Kin[3]  = { 64, 256, 256 };   // GEMM K per layer
    const int Fout[3] = { 64, 64, 16 };     // per-head output width
    const int HFo[3]  = { 256, 256, 64 };   // H * Fout

    // ---- workspace carve-up (all 256B aligned) ----
    char* ws = (char*)d_ws;
    size_t off = 0;
    auto carve = [&](size_t bytes) -> void* {
        void* p = ws + off;
        off = (off + bytes + 255) & ~(size_t)255;
        return p;
    };
    __bf16*   bfX     = (__bf16*)  carve((size_t)N * 256 * sizeof(__bf16));
    __bf16*   bfW     = (__bf16*)  carve((size_t)256 * 256 * sizeof(__bf16));
    float*    Xp      = (float*)   carve((size_t)N * 256 * sizeof(float));
    float*    outBuf  = (float*)   carve((size_t)N * 256 * sizeof(float));
    float*    aL      = (float*)   carve((size_t)N * 4 * sizeof(float));
    float*    aR      = (float*)   carve((size_t)N * 4 * sizeof(float));
    unsigned* smaxkey = (unsigned*)carve((size_t)N * 4 * sizeof(unsigned));
    float*    denom   = (float*)   carve((size_t)N * 4 * sizeof(float));
    float*    sbuf    = (float*)   carve((size_t)E * 4 * sizeof(float));
    float*    eetab   = (float*)   carve(256);
    (void)ws_size; (void)n_in; (void)out_size;

    const int TB = 256;
    const int edgeBlocks = 2048;

    const float* layerIn = x;
    for (int L = 0; L < 3; ++L) {
        const int K = Kin[L], F = Fout[L], HF = HFo[L];

        // 1) bf16 conversions (activations + weights) for the WMMA GEMM
        {
            int n = N * K;
            cvt_f32_bf16<<<(n + TB - 1) / TB, TB, 0, stream>>>(layerIn, bfX, n);
            int nw = HF * K;
            cvt_f32_bf16<<<(nw + TB - 1) / TB, TB, 0, stream>>>(Wl[L], bfW, nw);
        }

        // 2) Xp = bfX @ bfW^T via v_wmma_f32_16x16x32_bf16
        {
            int tiles = (N >> 4) * (HF >> 5);
            int blocks = (tiles + (TB >> 5) - 1) / (TB >> 5);
            gemm_wmma_bf16<<<blocks, TB, 0, stream>>>(bfX, bfW, Xp, N, K, HF);
        }

        // 3) tiny per-etype table + per-node attention dots
        ee_table<<<1, 32, 0, stream>>>(Eel[L], Wel[L], ael[L], eetab, ET);
        {
            int n = N * 4;
            node_attn_dots<<<(n + TB - 1) / TB, TB, 0, stream>>>(Xp, all[L], arl[L],
                                                                 aL, aR, N, F);
        }

        // 4) zero accumulators (key 0 == "below every real float")
        hipMemsetAsync(smaxkey, 0, (size_t)N * 4 * sizeof(unsigned), stream);
        hipMemsetAsync(denom,   0, (size_t)N * 4 * sizeof(float),    stream);
        hipMemsetAsync(outBuf,  0, (size_t)N * HF * sizeof(float),   stream);

        // 5) edge phase (L2-resident gather/scatter)
        edge_scores <<<edgeBlocks, TB, 0, stream>>>(row, col, et, aL, aR, eetab,
                                                    sbuf, smaxkey, E);
        edge_expsum <<<edgeBlocks, TB, 0, stream>>>(col, sbuf, smaxkey, denom, E);
        edge_scatter<<<edgeBlocks, TB, 0, stream>>>(row, col, sbuf, denom, Xp,
                                                    outBuf, E, F);

        // 6) activation
        if (L < 2) {
            int n = N * HF;
            relu_k<<<(n + TB - 1) / TB, TB, 0, stream>>>(outBuf, n);
        }
        layerIn = outBuf;
    }

    // 7) head-mean + L2 normalize -> d_out [N,16]
    final_norm<<<(N + TB - 1) / TB, TB, 0, stream>>>(outBuf, (float*)d_out, N);
}
